// Attention_89060441850459
// MI455X (gfx1250) — matrix operand order repaired and checked by reading, not compiled, not run on hardware
//
#include <hip/hip_runtime.h>

typedef __attribute__((ext_vector_type(16))) __bf16 v16bf;
typedef __attribute__((ext_vector_type(8)))  __bf16 v8bf;
typedef __attribute__((ext_vector_type(8)))  float  v8f;

#define BATCH  8
#define SEQ    1024
#define DMODEL 768
#define NHEAD  12
#define HDIM   64
#define D3     2304
#define TOK    (BATCH*SEQ)

#define WMMA_BF16(a,b,c) \
  __builtin_amdgcn_wmma_f32_16x16x32_bf16(false,(a),false,(b),(short)0,(c),false,false)

__device__ __forceinline__ v16bf cat8(v8bf lo, v8bf hi) {
  return __builtin_shufflevector(lo, hi, 0,1,2,3,4,5,6,7,8,9,10,11,12,13,14,15);
}

// Async copy: 16 bytes per lane, global -> LDS, tracked by ASYNCcnt (CDNA5).
// GVS mode: mem_addr = SGPR64 + VGPR32-offset; LDS dest addr comes from VGPR.
__device__ __forceinline__ void async_ld16(uint32_t lds_off, const void* sbase,
                                           uint32_t voff) {
  asm volatile("global_load_async_to_lds_b128 %0, %1, %2"
               :: "v"(lds_off), "v"(voff), "s"(sbase) : "memory");
}
__device__ __forceinline__ void wait_async0() {
  asm volatile("s_wait_asynccnt 0" ::: "memory");
}

// ---------------------------------------------------------------------------
// Kernel 1: qkv = X @ W_qkv  (f32 in -> bf16 WMMA -> bf16 Q/K/Vt scatter)
// block: 256 threads (8 waves), tile 128(M) x 128(N), k-step 32
// ---------------------------------------------------------------------------
#define LDA 40  // padded LDS row stride in bf16 elems (80B: 16B-aligned, conflict-free)

__global__ __launch_bounds__(256)
void qkv_gemm(const float* __restrict__ X, const float* __restrict__ W,
              __bf16* __restrict__ Qb, __bf16* __restrict__ Kb,
              __bf16* __restrict__ Vt)
{
  __shared__ __bf16 sA[128 * LDA];
  __shared__ __bf16 sB[128 * LDA];

  const int tid  = threadIdx.x;
  const int lane = tid & 31;
  const int wave = tid >> 5;
  const int wm   = wave >> 2;   // 0..1  (M direction, 64 rows each)
  const int wn   = wave & 3;    // 0..3  (N direction, 32 cols each)
  const int bn   = blockIdx.x;  // 0..17
  const int bm   = blockIdx.y;  // 0..63
  const int hi   = lane >> 4;   // lane half
  const int l16  = lane & 15;

  v8f acc[4][2];
  const v8f zero = {0.f,0.f,0.f,0.f,0.f,0.f,0.f,0.f};
  for (int i = 0; i < 4; ++i) for (int j = 0; j < 2; ++j) acc[i][j] = zero;

  for (int ks = 0; ks < DMODEL/32; ++ks) {
    const int k0 = ks * 32;
    __syncthreads();
    {   // A tile: 128x32 f32 -> bf16 (coalesced float4 reads)
      const int row = tid >> 3;
      const int cg  = (tid & 7) * 4;
      for (int i = 0; i < 4; ++i) {
        const int r = row + i*32;
        const float4 v = *(const float4*)(X + (size_t)(bm*128 + r)*DMODEL + k0 + cg);
        __bf16* d = &sA[r*LDA + cg];
        d[0] = (__bf16)v.x; d[1] = (__bf16)v.y; d[2] = (__bf16)v.z; d[3] = (__bf16)v.w;
      }
    }
    {   // B tile transposed into sB[n][k]
      const int col = tid & 127;
      const int kr  = tid >> 7;
      for (int i = 0; i < 16; ++i) {
        const int k = kr + i*2;
        sB[col*LDA + k] = (__bf16)W[(size_t)(k0 + k)*D3 + bn*128 + col];
      }
    }
    __syncthreads();

    v16bf af[4], bf2[2];
    for (int fm = 0; fm < 4; ++fm) {
      const __bf16* p = &sA[(wm*64 + fm*16 + l16)*LDA];
      af[fm] = cat8(*(const v8bf*)(p + hi*8), *(const v8bf*)(p + 16 + hi*8));
    }
    for (int fn = 0; fn < 2; ++fn) {
      const __bf16* p = &sB[(wn*32 + fn*16 + l16)*LDA];
      bf2[fn] = cat8(*(const v8bf*)(p + hi*8), *(const v8bf*)(p + 16 + hi*8));
    }
    for (int fm = 0; fm < 4; ++fm)
      for (int fn = 0; fn < 2; ++fn)
        acc[fm][fn] = WMMA_BF16(af[fm], bf2[fn], acc[fm][fn]);
  }

  // epilogue: scatter to Q (scaled by hd^-0.5), K, and V-transposed, all bf16
  for (int fm = 0; fm < 4; ++fm)
    for (int fn = 0; fn < 2; ++fn)
      for (int v = 0; v < 8; ++v) {
        const int m   = bm*128 + wm*64 + fm*16 + v + hi*8;   // token index
        const int c   = bn*128 + wn*32 + fn*16 + l16;        // 0..2303
        const int m3  = c / DMODEL;
        const int rem = c - m3*DMODEL;
        const int h   = rem >> 6;
        const int d   = rem & 63;
        const int b   = m >> 10;
        const int n   = m & 1023;
        const size_t bh = (size_t)b*NHEAD + h;
        const float val = acc[fm][fn][v];
        if (m3 == 0)      Qb[(bh*SEQ + n)*HDIM + d] = (__bf16)(val * 0.125f);
        else if (m3 == 1) Kb[(bh*SEQ + n)*HDIM + d] = (__bf16)val;
        else              Vt[(bh*HDIM + d)*SEQ + n] = (__bf16)val;
      }
}

// ---------------------------------------------------------------------------
// Kernel 2: flash attention, one (b,h) x 64 query rows per block (4 waves),
// each wave owns a 16-row WMMA M-tile; online softmax; double-buffered 64-key
// K/V chunks staged with global_load_async_to_lds_b128 (ASYNCcnt).
// ---------------------------------------------------------------------------
#define LDK 72  // padded stride (144B: 16B-aligned, conflict-free)

__device__ __forceinline__ void stage_kv(int tid, const __bf16* Kbase,
                                         const __bf16* Vbase, int kv0,
                                         __bf16* bK, __bf16* bV) {
  for (int i = 0; i < 4; ++i) {
    const int ch = tid + i*128;      // 0..511
    const int r  = ch >> 3;          // 0..63
    const int p8 = (ch & 7) * 8;     // 0..56
    async_ld16((uint32_t)(uintptr_t)&bK[r*LDK + p8], Kbase,
               (uint32_t)(((kv0 + r)*HDIM + p8) * 2u));
    async_ld16((uint32_t)(uintptr_t)&bV[r*LDK + p8], Vbase,
               (uint32_t)((r*SEQ + kv0 + p8) * 2u));
  }
}

__global__ __launch_bounds__(128)
void attn_kernel(const __bf16* __restrict__ Qb, const __bf16* __restrict__ Kb,
                 const __bf16* __restrict__ Vt, __bf16* __restrict__ AO)
{
  __shared__ __bf16 sK[2][64 * LDK];     // [key][dim], double-buffered
  __shared__ __bf16 sV[2][64 * LDK];     // [dim][key] (pre-transposed), dbl-buf
  __shared__ __bf16 sP[4][16 * LDK];     // per-wave C->A layout bounce

  const int tid  = threadIdx.x;
  const int lane = tid & 31;
  const int wave = tid >> 5;
  const int hi   = lane >> 4;
  const int l16  = lane & 15;
  const int bh   = blockIdx.y;                   // 0..95
  const int q0   = blockIdx.x*64 + wave*16;      // this wave's query rows

  const __bf16* Kbase = Kb + (size_t)bh*SEQ*HDIM;
  const __bf16* Vbase = Vt + (size_t)bh*HDIM*SEQ;

  // Q fragments (16x64 = two 16x32 A-frags), kept in registers
  v16bf qa[2];
  {
    const __bf16* qrow = Qb + ((size_t)bh*SEQ + q0 + l16)*HDIM;
    for (int kb = 0; kb < 2; ++kb)
      qa[kb] = cat8(*(const v8bf*)(qrow + kb*32 + hi*8),
                    *(const v8bf*)(qrow + kb*32 + 16 + hi*8));
  }

  float rmax[8], rsum[8];
  v8f oacc[4];
  const v8f zero = {0.f,0.f,0.f,0.f,0.f,0.f,0.f,0.f};
  for (int v = 0; v < 8; ++v) { rmax[v] = -1e30f; rsum[v] = 0.f; }
  for (int nt = 0; nt < 4; ++nt) oacc[nt] = zero;

  stage_kv(tid, Kbase, Vbase, 0, sK[0], sV[0]);   // prefetch chunk 0

  for (int j = 0; j < SEQ/64; ++j) {
    const int cur = j & 1;
    wait_async0();          // my loads for buffer `cur` have landed
    __syncthreads();        // everyone's have; buffer cur^1 free to overwrite
    if (j + 1 < SEQ/64)
      stage_kv(tid, Kbase, Vbase, (j+1)*64, sK[cur^1], sV[cur^1]);

    const __bf16* cK = sK[cur];
    const __bf16* cV = sV[cur];

    // S = Q @ K^T  -> 16 x 64 scores (4 C-frags)
    v8f s[4];
    for (int nt = 0; nt < 4; ++nt) {
      s[nt] = zero;
      const __bf16* kp = &cK[(nt*16 + l16)*LDK];
      for (int kb = 0; kb < 2; ++kb) {
        v16bf bk = cat8(*(const v8bf*)(kp + kb*32 + hi*8),
                        *(const v8bf*)(kp + kb*32 + 16 + hi*8));
        s[nt] = WMMA_BF16(qa[kb], bk, s[nt]);
      }
    }

    // online softmax: row stats live per (vgpr, lane-half); reduce over 16 lanes
    float csum[8];
    for (int v = 0; v < 8; ++v) {
      float m = fmaxf(fmaxf(s[0][v], s[1][v]), fmaxf(s[2][v], s[3][v]));
      for (int off = 1; off < 16; off <<= 1)
        m = fmaxf(m, __shfl_xor(m, off, 32));
      const float nm = fmaxf(rmax[v], m);
      const float rescale = __expf(rmax[v] - nm);
      rmax[v] = nm;
      rsum[v] *= rescale;
      for (int nt = 0; nt < 4; ++nt) oacc[nt][v] *= rescale;
      float cs = 0.f;
      for (int nt = 0; nt < 4; ++nt) {
        const float p = __expf(s[nt][v] - nm);
        s[nt][v] = p;
        cs += p;
      }
      csum[v] = cs;
    }
    for (int v = 0; v < 8; ++v) {
      float cs = csum[v];
      for (int off = 1; off < 16; off <<= 1)
        cs += __shfl_xor(cs, off, 32);
      rsum[v] += cs;
    }

    // C-layout -> A-layout via wave-private LDS bounce (bf16)
    __bf16* pb = sP[wave];
    for (int nt = 0; nt < 4; ++nt)
      for (int v = 0; v < 8; ++v)
        pb[(v + hi*8)*LDK + nt*16 + l16] = (__bf16)s[nt][v];
    asm volatile("s_wait_dscnt 0" ::: "memory");   // wave-local RAW through LDS

    // O += P @ V
    for (int kb = 0; kb < 2; ++kb) {
      const __bf16* pr = &pb[l16*LDK + kb*32];
      v16bf pa = cat8(*(const v8bf*)(pr + hi*8), *(const v8bf*)(pr + 16 + hi*8));
      for (int nt = 0; nt < 4; ++nt) {
        const __bf16* vp = &cV[(nt*16 + l16)*LDK + kb*32];
        v16bf bv = cat8(*(const v8bf*)(vp + hi*8), *(const v8bf*)(vp + 16 + hi*8));
        oacc[nt] = WMMA_BF16(pa, bv, oacc[nt]);
      }
    }
  }

  // normalize + store merged heads: AO[b][n][h*64+d] (bf16)
  const int b = bh / NHEAD;
  const int h = bh - b*NHEAD;
  for (int nt = 0; nt < 4; ++nt)
    for (int v = 0; v < 8; ++v) {
      const int m = v + hi*8;
      const size_t row = (size_t)b*SEQ + q0 + m;
      const int col = h*HDIM + nt*16 + l16;
      AO[row*DMODEL + col] = (__bf16)(oacc[nt][v] / rsum[v]);
    }
}

// ---------------------------------------------------------------------------
// Kernel 3: out = AO @ W_proj + b_proj   (bf16 A async-staged, f32 W, f32 out)
// ---------------------------------------------------------------------------
__global__ __launch_bounds__(256)
void proj_gemm(const __bf16* __restrict__ A, const float* __restrict__ W,
               const float* __restrict__ bias, float* __restrict__ out)
{
  __shared__ __bf16 sA[128 * LDA];
  __shared__ __bf16 sB[128 * LDA];

  const int tid  = threadIdx.x;
  const int lane = tid & 31;
  const int wave = tid >> 5;
  const int wm   = wave >> 2;
  const int wn   = wave & 3;
  const int bn   = blockIdx.x;   // 0..5
  const int bm   = blockIdx.y;   // 0..63
  const int hi   = lane >> 4;
  const int l16  = lane & 15;

  const __bf16* Abase = A + (size_t)bm*128*DMODEL;

  v8f acc[4][2];
  const v8f zero = {0.f,0.f,0.f,0.f,0.f,0.f,0.f,0.f};
  for (int i = 0; i < 4; ++i) for (int j = 0; j < 2; ++j) acc[i][j] = zero;

  for (int ks = 0; ks < DMODEL/32; ++ks) {
    const int k0 = ks * 32;
    __syncthreads();
    {   // A tile: bf16, direct async copy to LDS (overlaps with B conversion)
      for (int i = 0; i < 2; ++i) {
        const int ch = tid + i*256;    // 0..511
        const int r  = ch >> 2;        // 0..127
        const int p8 = (ch & 3) * 8;   // 0..24
        async_ld16((uint32_t)(uintptr_t)&sA[r*LDA + p8], Abase,
                   (uint32_t)((r*DMODEL + k0 + p8) * 2u));
      }
    }
    {   // B tile transposed, f32 -> bf16 via VALU
      const int col = tid & 127;
      const int kr  = tid >> 7;
      for (int i = 0; i < 16; ++i) {
        const int k = kr + i*2;
        sB[col*LDA + k] = (__bf16)W[(size_t)(k0 + k)*DMODEL + bn*128 + col];
      }
    }
    wait_async0();
    __syncthreads();

    v16bf af[4], bf2[2];
    for (int fm = 0; fm < 4; ++fm) {
      const __bf16* p = &sA[(wm*64 + fm*16 + l16)*LDA];
      af[fm] = cat8(*(const v8bf*)(p + hi*8), *(const v8bf*)(p + 16 + hi*8));
    }
    for (int fn = 0; fn < 2; ++fn) {
      const __bf16* p = &sB[(wn*32 + fn*16 + l16)*LDA];
      bf2[fn] = cat8(*(const v8bf*)(p + hi*8), *(const v8bf*)(p + 16 + hi*8));
    }
    for (int fm = 0; fm < 4; ++fm)
      for (int fn = 0; fn < 2; ++fn)
        acc[fm][fn] = WMMA_BF16(af[fm], bf2[fn], acc[fm][fn]);
  }

  for (int fm = 0; fm < 4; ++fm)
    for (int fn = 0; fn < 2; ++fn)
      for (int v = 0; v < 8; ++v) {
        const int m = bm*128 + wm*64 + fm*16 + v + hi*8;
        const int c = bn*128 + wn*32 + fn*16 + l16;
        out[(size_t)m*DMODEL + c] = acc[fm][fn][v] + bias[c];
      }
}

// ---------------------------------------------------------------------------
extern "C" void kernel_launch(void* const* d_in, const int* in_sizes, int n_in,
                              void* d_out, int out_size, void* d_ws, size_t ws_size,
                              hipStream_t stream)
{
  (void)in_sizes; (void)n_in; (void)out_size; (void)ws_size;
  const float* X  = (const float*)d_in[0];   // [8,1024,768]
  const float* Wq = (const float*)d_in[1];   // [768,2304]
  const float* Wp = (const float*)d_in[2];   // [768,768]
  const float* bp = (const float*)d_in[3];   // [768]
  float* out = (float*)d_out;

  const size_t SZ = (size_t)BATCH*NHEAD*SEQ*HDIM;  // 6,291,456 elems
  __bf16* Qb = (__bf16*)d_ws;      // [b][h][n][d], pre-scaled
  __bf16* Kb = Qb + SZ;            // [b][h][n][d]
  __bf16* Vt = Kb + SZ;            // [b][h][d][n]
  __bf16* AO = Vt + SZ;            // [token][768]

  qkv_gemm <<<dim3(D3/128,     TOK/128),   256, 0, stream>>>(X, Wq, Qb, Kb, Vt);
  attn_kernel<<<dim3(SEQ/64, BATCH*NHEAD), 128, 0, stream>>>(Qb, Kb, Vt, AO);
  proj_gemm<<<dim3(DMODEL/128, TOK/128),   256, 0, stream>>>(AO, Wp, bp, out);
}
